// RNN1DGeneral_15607911153978
// MI455X (gfx1250) — matrix operand order repaired and checked by reading, not compiled, not run on hardware
//
#include <hip/hip_runtime.h>
#include <hip/hip_bf16.h>

// Problem constants (fixed by the reference)
#define BB     256
#define LL     4096
#define HH     128
#define HPAD   136   // 272B row stride = 68 dwords -> conflict-free ds_load_b128
#define MTILE  16
#define CHUNK  256   // x-timestep staging chunk in LDS

typedef __bf16 bf16_t;
typedef bf16_t v16bf __attribute__((ext_vector_type(16)));
typedef bf16_t v8bf  __attribute__((ext_vector_type(8)));
typedef float  v8f   __attribute__((ext_vector_type(8)));

union bfreg_u { v16bf b; v8f f; };

__device__ __forceinline__ float elu_f(float x) {
    return x > 0.f ? x : (__expf(x) - 1.f);
}

// One block = one WMMA M-tile of 16 batch rows. 8 waves; wave w owns hidden
// columns [16w, 16w+16). Weight (B) fragments pinned in bf16 VGPRs for all
// 4096 steps; recurrent state double-buffered in LDS as bf16. A-fragment
// loads are software-pipelined one step ahead (issued right after B1, consumed
// next step) so DS latency overlaps the output head + barriers.
__global__ __launch_bounds__(256) void rnn1d_seq_kernel(
    const int*   __restrict__ d_x,     // [B, L] int32
    const float* __restrict__ d_Win,   // [2, 128]
    const float* __restrict__ d_Wc,    // [2, 128, 128]
    const float* __restrict__ d_Wout,  // [128, 1]
    const float* __restrict__ d_bout,  // [1]
    float*       __restrict__ d_out)   // [B]
{
    __shared__ bf16_t h0s[2][MTILE][HPAD];
    __shared__ bf16_t h1s[2][MTILE][HPAD];
    __shared__ int    xbuf[MTILE][CHUNK];   // staged x chunk
    __shared__ int    xmask[CHUNK];         // bit b = x[b][t] for this chunk
    __shared__ int    xcarrm;               // xmask[CHUNK-1] of previous chunk
    __shared__ float  part[MTILE][17];

    const int tid = threadIdx.x;
    const int w   = tid >> 5;       // wave id (0..7)
    const int l   = tid & 31;       // lane
    const int hi  = l >> 4;         // half-wave
    const int nn  = l & 15;
    const int cN  = w * 16 + nn;    // hidden column owned by this lane (C layout)
    const int m   = nn;             // A-fragment row (batch row within tile)
    const int b0  = blockIdx.x * MTILE;

    // ---- zero-init recurrent state (zero_carry), both buffers ----
    for (int i = tid; i < 2 * MTILE * HPAD; i += 256) {
        ((bf16_t*)h0s)[i] = (bf16_t)0.f;
        ((bf16_t*)h1s)[i] = (bf16_t)0.f;
    }

    // ---- pin B fragments (W_c, bf16) in registers: 2 layers x 4 k-steps ----
    // B 32x16 layout: lane%16 = N, elems 0..7 -> K = 32s + 8*(lane/16) + e,
    //                 elems 8..15 -> K = 32s + 16 + 8*(lane/16) + (e-8)
    const int koff = hi * 8;
    v16bf Bf[2][4];
    #pragma unroll
    for (int j = 0; j < 2; ++j)
      #pragma unroll
      for (int s = 0; s < 4; ++s) {
        bfreg_u u;
        #pragma unroll
        for (int e = 0; e < 16; ++e) {
            const int k = 32 * s + koff + (e < 8 ? e : e + 8);
            u.b[e] = (bf16_t)d_Wc[(j * HH + k) * HH + cN];
        }
        // Opaque register boundary: keeps the bf16 tuple resident in VGPRs
        // across the whole time loop (stops per-iteration re-conversion).
        asm volatile("" : "+v"(u.f));
        Bf[j][s] = u.b;
      }

    // r = elu(prev @ W_in): prev is one-hot over 2 classes -> 2-entry table
    const float r0 = elu_f(d_Win[cN]);
    const float r1 = elu_f(d_Win[HH + cN]);

    // ---- output-head per-thread constants ----
    const int hb  = tid >> 4;       // batch row for head partial
    const int seg = tid & 15;       // 8-column segment
    float wreg[8];
    #pragma unroll
    for (int i = 0; i < 8; ++i) wreg[i] = d_Wout[seg * 8 + i];
    const float bout = d_bout[0];

    float acc = 0.f;                // per-sample log-prob (tid < 16 only)

    __syncthreads();

    // A 16x32 layout: lane row = m, elems 0..7 -> K = kb + 8*hi + e,
    //                 elems 8..15 -> K = kb + 16 + 8*hi + (e-8)
    auto loadA = [&](int buf, v16bf* A0, v16bf* A1) {
        #pragma unroll
        for (int s = 0; s < 4; ++s) {
            const int kb = 32 * s;
            v8bf lo0 = *reinterpret_cast<const v8bf*>(&h0s[buf][m][kb + hi * 8]);
            v8bf up0 = *reinterpret_cast<const v8bf*>(&h0s[buf][m][kb + 16 + hi * 8]);
            A0[s] = __builtin_shufflevector(lo0, up0, 0,1,2,3,4,5,6,7,8,9,10,11,12,13,14,15);
            v8bf lo1 = *reinterpret_cast<const v8bf*>(&h1s[buf][m][kb + hi * 8]);
            v8bf up1 = *reinterpret_cast<const v8bf*>(&h1s[buf][m][kb + 16 + hi * 8]);
            A1[s] = __builtin_shufflevector(lo1, up1, 0,1,2,3,4,5,6,7,8,9,10,11,12,13,14,15);
        }
    };

    // One timestep. cA*: fragments preloaded for this step; nA*: fragments for
    // the next step, loaded here right after B1 (overlaps head + B2 latency).
    auto body = [&](int t, v16bf* cA0, v16bf* cA1, v16bf* nA0, v16bf* nA1) {
        const int lt = t & (CHUNK - 1);
        const int rd = t & 1;       // state buffer read this step
        const int wr = rd ^ 1;      // state buffer written this step

        // ---- stage next x chunk + pack per-step 16-bit batch masks ----
        if (lt == 0) {
            if (t > 0 && tid == 0) xcarrm = xmask[CHUNK - 1];
            __syncthreads();
            const int* xrow = d_x + (size_t)(b0 + hb) * LL + t;
            #pragma unroll
            for (int q = 0; q < (CHUNK / 16) / 4; ++q) {
                int4 v = *reinterpret_cast<const int4*>(xrow + seg * (CHUNK / 16) + q * 4);
                *reinterpret_cast<int4*>(&xbuf[hb][seg * (CHUNK / 16) + q * 4]) = v;
            }
            __syncthreads();
            int mval = 0;
            #pragma unroll
            for (int b = 0; b < MTILE; ++b) mval |= (xbuf[b][tid] & 1) << b;
            xmask[tid] = mval;
            __syncthreads();
        }

        // ---- both layers' GEMMs (inputs are prev-step carry; A preloaded) ----
        v8f C0 = {}, C1 = {};
        #pragma unroll
        for (int s = 0; s < 4; ++s) {
            C0 = __builtin_amdgcn_wmma_f32_16x16x32_bf16(false, cA0[s], false, Bf[0][s],
                                                         (short)0, C0, false, false);
            C1 = __builtin_amdgcn_wmma_f32_16x16x32_bf16(false, cA1[s], false, Bf[1][s],
                                                         (short)0, C1, false, false);
        }

        // ---- r gate: one broadcast LDS read + bit extracts ----
        float rv[8];
        if (t == 0) {
            #pragma unroll
            for (int v = 0; v < 8; ++v) rv[v] = 0.f;   // prev0 = zeros -> elu(0) = 0
        } else {
            const int xm = (lt == 0) ? xcarrm : xmask[lt - 1];
            #pragma unroll
            for (int v = 0; v < 8; ++v) {
                const int b = v + 8 * hi;
                rv[v] = ((xm >> b) & 1) ? r1 : r0;
            }
        }

        // ---- elementwise: h0' = elu(C0)+r ; h1' = elu(C1)+h0' ; store bf16 ----
        #pragma unroll
        for (int v = 0; v < 8; ++v) {
            const int b = v + 8 * hi;          // C layout: VGPR v -> M = v + 8*hi
            const float h0v = elu_f(C0[v]) + rv[v];
            const float h1v = elu_f(C1[v]) + h0v;
            h0s[wr][b][cN] = (bf16_t)h0v;
            h1s[wr][b][cN] = (bf16_t)h1v;
        }
        __syncthreads();   // B1: new state visible to all waves

        // ---- preload next step's A fragments (state just written to wr) ----
        // Live across B2 into the next body -> allocator must keep all 16
        // b128 loads in distinct registers; latency hidden behind head work.
        loadA(wr, nA0, nA1);

        // ---- output head: logit[b] = h1'[b,:] . W_out + b_out ----
        v8bf hv = *reinterpret_cast<const v8bf*>(&h1s[wr][hb][seg * 8]);
        float p = 0.f;
        #pragma unroll
        for (int i = 0; i < 8; ++i) p += (float)hv[i] * wreg[i];
        part[hb][seg] = p;
        __syncthreads();   // B2: partials visible

        if (tid < MTILE) {
            float logit = bout;
            #pragma unroll
            for (int k = 0; k < 16; ++k) logit += part[tid][k];
            const int xc = (xmask[lt] >> tid) & 1;
            // 0.5 * log_softmax([0, logit]) at class xc, numerically stable
            const float lsum = fmaxf(logit, 0.f) + __logf(1.f + __expf(-fabsf(logit)));
            float lp = 0.5f * ((xc ? logit : 0.f) - lsum);
            if (lp != lp) lp = -35.f;          // nan_to_num
            acc += lp;
        }
    };

    // Ping-pong fragment sets; 2x manual unroll avoids cross-step reg copies.
    v16bf Aa0[4], Aa1[4], Ab0[4], Ab1[4];
    loadA(0, Aa0, Aa1);   // step 0 reads the zeroed buffer 0

    for (int t2 = 0; t2 < LL; t2 += 2) {
        body(t2,     Aa0, Aa1, Ab0, Ab1);
        body(t2 + 1, Ab0, Ab1, Aa0, Aa1);
    }

    if (tid < MTILE) d_out[b0 + tid] = acc;
}

extern "C" void kernel_launch(void* const* d_in, const int* in_sizes, int n_in,
                              void* d_out, int out_size, void* d_ws, size_t ws_size,
                              hipStream_t stream) {
    (void)in_sizes; (void)n_in; (void)out_size; (void)d_ws; (void)ws_size;
    const int*   d_x    = (const int*)  d_in[0];
    const float* d_Win  = (const float*)d_in[1];
    const float* d_Wc   = (const float*)d_in[2];
    const float* d_Wout = (const float*)d_in[3];
    const float* d_bout = (const float*)d_in[4];
    float*       out    = (float*)d_out;

    // 16 blocks (one per 16-row batch tile) x 256 threads (8 wave32s)
    rnn1d_seq_kernel<<<dim3(BB / MTILE), dim3(256), 0, stream>>>(
        d_x, d_Win, d_Wc, d_Wout, d_bout, out);
}
